// _Conv_24429773980100
// MI455X (gfx1250) — matrix operand order repaired and checked by reading, not compiled, not run on hardware
//
#include <hip/hip_runtime.h>
#include <hip/hip_bf16.h>

typedef __attribute__((ext_vector_type(16))) _Float16 v16h;
typedef __attribute__((ext_vector_type(8)))  _Float16 v8h;
typedef __attribute__((ext_vector_type(4)))  _Float16 v4h;
typedef __attribute__((ext_vector_type(8)))  float    v8f;
typedef __attribute__((ext_vector_type(4)))  float    v4f;
typedef __attribute__((ext_vector_type(4)))  int      v4i;

#define EPSILON_ 0.001f

// Problem dims
#define H_    56
#define W_    56
#define C_    256
#define CO_   256
#define KTOT  2304    // 3*3*256
#define HWPIX 3136    // 56*56
#define MTOT  100352  // 32*3136

// Tiling: block = 64 pixels x 256 out-channels, K-step 32.
#define BM 64
#define BN 256
#define BK 32
#define LDS_STRIDE 40   // halfs per LDS row (80B): conflict-free b128 frag loads
#define A_BUF 4096      // halfs per A buffer (8KB, pow2 -> cheap ping-pong)
#define B_BUF 8192      // halfs per B buffer (16KB, pow2)

// ---- CDNA5 async global->LDS path (ASYNCcnt), guarded for portability ----
#if defined(__has_builtin)
#  if __has_builtin(__builtin_amdgcn_global_load_async_to_lds_b128)
#    define HAVE_ASYNC_LDS 1
#  endif
#endif
#ifndef HAVE_ASYNC_LDS
#  define HAVE_ASYNC_LDS 0
#endif

#if HAVE_ASYNC_LDS
// Probe-learned signature: (v4i AS1*, v4i AS3*, imm offset, imm cpol)
typedef __attribute__((address_space(1))) v4i GAS1_v4i;
typedef __attribute__((address_space(3))) v4i LAS3_v4i;

__device__ __forceinline__ void async_copy_b128(const void* g, void* l) {
    __builtin_amdgcn_global_load_async_to_lds_b128(
        (GAS1_v4i*)(g),
        (LAS3_v4i*)(l),
        0, 0);
}
__device__ __forceinline__ void wait_async0() {
#if defined(__has_builtin) && __has_builtin(__builtin_amdgcn_s_wait_asynccnt)
    __builtin_amdgcn_s_wait_asynccnt(0);
#else
    asm volatile("s_wait_asynccnt 0x0" ::: "memory");
#endif
}
#endif

// ---------------------------------------------------------------------------
// Prepass: soft-binarize weights, convert to f16, transpose HWIO[k][co]
// -> wT[co][k] so B-tiles stream into LDS as contiguous 64B rows.
// ---------------------------------------------------------------------------
__global__ void binarize_weights(const float* __restrict__ w,
                                 _Float16* __restrict__ wT) {
    int idx = blockIdx.x * blockDim.x + threadIdx.x;   // k*CO_ + co
    if (idx >= KTOT * CO_) return;
    int k  = idx / CO_;
    int co = idx - k * CO_;
    float v = w[idx];
    float b = v * (1.0f + EPSILON_) / (fabsf(v) + EPSILON_);
    wT[(size_t)co * KTOT + k] = (_Float16)b;
}

// ---------------------------------------------------------------------------
// Implicit-GEMM conv: M=100352 pixels, N=256 channels, K=2304.
// 256 threads = 8 waves in 2(M) x 4(N); wave tile 32x64 = 2x4 WMMAs.
// Double-buffered LDS; A: register prefetch + f32->f16 convert;
// B: GLOBAL_LOAD_ASYNC_TO_LDS when available, register prefetch otherwise.
// ---------------------------------------------------------------------------
__global__ __launch_bounds__(256)
void conv3x3_wmma(const float* __restrict__ x,
                  const _Float16* __restrict__ wT,
                  float* __restrict__ out) {
    __shared__ _Float16 As[2 * A_BUF];
    __shared__ _Float16 Bs[2 * B_BUF];

    const int tid   = threadIdx.x;
    const int lane  = tid & 31;
    const int wave  = tid >> 5;
    const int waveM = wave >> 2;        // 0..1
    const int waveN = wave & 3;         // 0..3
    const int mBase = blockIdx.x * BM;  // 1568 blocks

    // ---- A staging geometry: 2 rows/thread, fixed 4-float quad ----
    const int qA = tid & 7;
    int rowA[2], npix[2], ph[2], pw[2];
#pragma unroll
    for (int i = 0; i < 2; ++i) {
        int row = (tid >> 3) + 32 * i;
        rowA[i] = row;
        int p  = mBase + row;
        int n  = p / HWPIX;
        int hw = p - n * HWPIX;
        int h  = hw / W_;
        npix[i] = n * HWPIX;
        ph[i]   = h;
        pw[i]   = hw - h * W_;
    }
    // ---- B staging geometry: 4 rows/thread, fixed 8-half chunk ----
    const int qB = tid & 3;
    int rowB[4];
    unsigned bOff[4];
#pragma unroll
    for (int i = 0; i < 4; ++i) {
        int row = (tid >> 2) + 64 * i;
        rowB[i] = row;
        bOff[i] = (unsigned)(row * KTOT + qB * 8) * 2u;   // bytes into wT
    }

    v8f acc[2][4] = {};                 // f32 accumulators
    const int aKb = (lane < 16) ? 0 : 8;
    const int bKb = (lane < 16) ? 0 : 8;
    const int lr  = lane & 15;
    const int mHalf = (lane < 16) ? 0 : 8;

    unsigned aOff[2], aMask[2];
    v4f av[2];                          // prefetched A regs (f32)
#if !HAVE_ASYNC_LDS
    v8h bv[4];                          // prefetched B regs (f16)
#endif

    // Tap rebase: clamped (always in-bounds) offset + zero-mask for the halo.
    auto rebase = [&](int tap) {
        int dh = tap / 3 - 1;
        int dw = tap - (tap / 3) * 3 - 1;
#pragma unroll
        for (int i = 0; i < 2; ++i) {
            int h2 = ph[i] + dh, w2 = pw[i] + dw;
            bool ok = ((unsigned)h2 < (unsigned)H_) & ((unsigned)w2 < (unsigned)W_);
            int h2c = ok ? h2 : 0;
            int w2c = ok ? w2 : 0;
            aOff[i]  = (unsigned)((npix[i] + h2c * W_ + w2c) * C_ + qA * 4) * 4u;
            aMask[i] = ok ? 0xFFFFFFFFu : 0u;
        }
    };
    auto fetchA = [&]() {
#pragma unroll
        for (int i = 0; i < 2; ++i) {
            av[i] = *(const v4f*)((const char*)x + aOff[i]);
            aOff[i] += BK * 4;          // next 32-channel slab: +128B
        }
    };
    auto fetchB = [&](int dstBuf) {
#if HAVE_ASYNC_LDS
        _Float16* bB = &Bs[dstBuf * B_BUF];
#pragma unroll
        for (int i = 0; i < 4; ++i) {
            async_copy_b128((const char*)wT + bOff[i],
                            &bB[rowB[i] * LDS_STRIDE + qB * 8]);
            bOff[i] += BK * 2;          // +64B along K
        }
#else
        (void)dstBuf;
#pragma unroll
        for (int i = 0; i < 4; ++i) {
            bv[i] = *(const v8h*)((const char*)wT + bOff[i]);
            bOff[i] += BK * 2;          // +64B along K
        }
#endif
    };

    rebase(0);
    fetchA();
    fetchB(0);                          // async path: kb=0 -> LDS buf 0

    int buf = 0;
    for (int kb = 0; kb < KTOT / BK; ++kb) {     // 72 K-blocks
        // ---- stage prefetched A regs -> LDS[buf] ----
        _Float16* aB = &As[buf * A_BUF];
        _Float16* bB = &Bs[buf * B_BUF];
#pragma unroll
        for (int i = 0; i < 2; ++i) {
            union { v4h h; unsigned u[2]; } cv;
            cv.h = (v4h){ (_Float16)av[i].x, (_Float16)av[i].y,
                          (_Float16)av[i].z, (_Float16)av[i].w };
            cv.u[0] &= aMask[i];
            cv.u[1] &= aMask[i];
            *(v4h*)&aB[rowA[i] * LDS_STRIDE + qA * 4] = cv.h;
        }
#if HAVE_ASYNC_LDS
        wait_async0();                  // B async writes to LDS[buf] visible
#else
#pragma unroll
        for (int i = 0; i < 4; ++i)
            *(v8h*)&bB[rowB[i] * LDS_STRIDE + qB * 8] = bv[i];
#endif
        __syncthreads();

        // ---- fragment loads from LDS[buf] (layouts per ISA 7.12.2) ----
        v16h afr[2];
#pragma unroll
        for (int mi = 0; mi < 2; ++mi) {
            const _Float16* ap = &aB[(waveM * 32 + mi * 16 + lr) * LDS_STRIDE];
            v8h lo = *(const v8h*)(ap + aKb);
            v8h hi = *(const v8h*)(ap + aKb + 16);
            afr[mi] = __builtin_shufflevector(lo, hi,
                0,1,2,3,4,5,6,7,8,9,10,11,12,13,14,15);
        }
        v16h bfr[4];
#pragma unroll
        for (int ni = 0; ni < 4; ++ni) {
            const _Float16* bp = &bB[(waveN * 64 + ni * 16 + lr) * LDS_STRIDE + bKb];
            v8h lo = *(const v8h*)(bp);
            v8h hi = *(const v8h*)(bp + 16);
            bfr[ni] = __builtin_shufflevector(lo, hi,
                0,1,2,3,4,5,6,7,8,9,10,11,12,13,14,15);
        }

        // ---- prefetch next K-block (latency hidden by WMMAs) ----
        if (kb + 1 < KTOT / BK) {
            if (((kb + 1) & 7) == 0) rebase((kb + 1) >> 3);
            fetchA();
            fetchB(buf ^ 1);            // async path writes LDS[buf^1] directly
        }

        // ---- 8 WMMAs: D = A x B + C, f32 accumulate ----
#pragma unroll
        for (int mi = 0; mi < 2; ++mi)
#pragma unroll
            for (int ni = 0; ni < 4; ++ni)
                acc[mi][ni] = __builtin_amdgcn_wmma_f32_16x16x32_f16(
                    false, afr[mi], false, bfr[ni],
                    (short)0, acc[mi][ni], false, false);

        __syncthreads();
        buf ^= 1;
    }

    // ---- epilogue: ReLU + f32 store (D layout: M = v + 8*(lane>=16)) ----
#pragma unroll
    for (int mi = 0; mi < 2; ++mi) {
        int m0 = mBase + waveM * 32 + mi * 16 + mHalf;
#pragma unroll
        for (int ni = 0; ni < 4; ++ni) {
            int col = waveN * 64 + ni * 16 + lr;
#pragma unroll
            for (int v = 0; v < 8; ++v) {
                float val = acc[mi][ni][v];
                val = val > 0.0f ? val : 0.0f;
                out[(size_t)(m0 + v) * CO_ + col] = val;
            }
        }
    }
}

// ---------------------------------------------------------------------------
extern "C" void kernel_launch(void* const* d_in, const int* in_sizes, int n_in,
                              void* d_out, int out_size, void* d_ws, size_t ws_size,
                              hipStream_t stream) {
    const float* x = (const float*)d_in[0];       // (32,56,56,256) f32
    const float* w = (const float*)d_in[1];       // (3,3,256,256) f32 HWIO
    float* out = (float*)d_out;                   // (32,56,56,256) f32
    _Float16* wT = (_Float16*)d_ws;               // 2304*256 f16 = 1.18 MB

    int total = KTOT * CO_;
    binarize_weights<<<(total + 255) / 256, 256, 0, stream>>>(w, wT);

    dim3 grid(MTOT / BM);                         // 1568 blocks, full Cout each
    conv3x3_wmma<<<grid, 256, 0, stream>>>(x, wT, out);
}
